// DecoderLayer_55825984914053
// MI455X (gfx1250) — matrix operand order repaired and checked by reading, not compiled, not run on hardware
//
#include <hip/hip_runtime.h>
#include <hip/hip_bf16.h>
#include <math.h>

// ---------------------------------------------------------------------------
// Deformable-DETR decoder layer for MI455X (gfx1250, wave32, WMMA).
// GEMMs: bf16 inputs (packed-converted on LDS stage-in), f32 accumulation via
// v_wmma_f32_16x16x32_bf16. Attention is flash-style with the S^T trick:
// S^T = K Q^T so the score D-layout coincides with the P A-fragment layout
// (no LDS transpose of P), per-lane online softmax, transposed V staging.
// ---------------------------------------------------------------------------

typedef __attribute__((ext_vector_type(16))) __bf16 v16bf;
typedef __attribute__((ext_vector_type(8)))  __bf16 v8bf;
typedef __attribute__((ext_vector_type(4)))  __bf16 v4bf;
typedef __attribute__((ext_vector_type(8)))  float  v8f;

#define NHEADS 8
#define DH     32
#define CDIM   256
#define BDIM   2
#define LQ     2304
#define LEN_IN 3060
#define DFF    1024
#define NLVL   4

static __device__ __forceinline__ __bf16 to_bf(float x) { return (__bf16)x; }

// ===========================================================================
// Generic GEMM: C[M,N] = (A1 (+A2)) @ W^T (+bias) (+res) (ReLU optional)
// A1/A2: (M,K) f32.  W: (N,K) f32 row-major (PyTorch Linear weight layout).
// Tile 64x64x32; 8 waves; wave -> 16 rows x 32 cols (2 WMMA accumulators).
// Stage-in: 8 contiguous elems/thread = 2x global_load_b128 -> 4x
// v_cvt_pk_bf16_f32 -> 16B LDS store (rows padded to 80B, 16B aligned).
// ===========================================================================
#define TM 64
#define TN 64
#define TK 32
#define LDP 40   // padded LDS row stride in bf16 (80B)

template<bool HAS_A2>
__global__ __launch_bounds__(256) void gemm_bf16_kernel(
    const float* __restrict__ A1, const float* __restrict__ A2,
    const float* __restrict__ W,  const float* __restrict__ bias,
    const float* __restrict__ res, float* __restrict__ Cout,
    int M, int N, int K, int relu)
{
    __shared__ alignas(16) __bf16 As[TM][LDP];
    __shared__ alignas(16) __bf16 Bs[TN][LDP];

    const int tid  = threadIdx.x;
    const int lane = tid & 31;
    const int wave = tid >> 5;
    const int wm   = wave >> 1;          // 0..3  (16-row sub-tile)
    const int wn   = wave & 1;           // 0..1  (32-col sub-tile)
    const int hf   = lane >> 4;          // wave half
    const int q16  = lane & 15;
    const int tm   = blockIdx.x * TM;
    const int tn   = blockIdx.y * TN;

    // staging coordinates: thread owns 8 consecutive K-elements of one row
    const int ar = tid >> 2;             // 0..63 (tile row)
    const int ac = (tid & 3) * 8;        // 0,8,16,24 (tile col)
    const int am = tm + ar;
    const bool arow_ok = (am < M);
    const size_t abase = (size_t)am * K + ac;
    const size_t bbase = (size_t)(tn + ar) * K + ac;

    v8f acc0 = {0.f,0.f,0.f,0.f,0.f,0.f,0.f,0.f};
    v8f acc1 = {0.f,0.f,0.f,0.f,0.f,0.f,0.f,0.f};

    for (int k0 = 0; k0 < K; k0 += TK) {
        // ---- stage A tile (vectorized, fused q = tgt + pos) ----
        float4 a0 = {0.f,0.f,0.f,0.f}, a1 = {0.f,0.f,0.f,0.f};
        if (arow_ok) {
            const float4* pa = (const float4*)(A1 + abase + k0);
            a0 = pa[0]; a1 = pa[1];
            if (HAS_A2) {
                const float4* p2 = (const float4*)(A2 + abase + k0);
                float4 c0 = p2[0], c1 = p2[1];
                a0.x += c0.x; a0.y += c0.y; a0.z += c0.z; a0.w += c0.w;
                a1.x += c1.x; a1.y += c1.y; a1.z += c1.z; a1.w += c1.w;
            }
            if (k0 + TK < K) __builtin_prefetch(A1 + abase + k0 + TK, 0, 1);
        }
        v8bf apk;
        apk[0] = to_bf(a0.x); apk[1] = to_bf(a0.y);
        apk[2] = to_bf(a0.z); apk[3] = to_bf(a0.w);
        apk[4] = to_bf(a1.x); apk[5] = to_bf(a1.y);
        apk[6] = to_bf(a1.z); apk[7] = to_bf(a1.w);
        *(v8bf*)&As[ar][ac] = apk;

        // ---- stage W tile (vectorized) ----
        const float4* pw = (const float4*)(W + bbase + k0);
        float4 b0 = pw[0], b1 = pw[1];
        if (k0 + TK < K) __builtin_prefetch(W + bbase + k0 + TK, 0, 1);
        v8bf bpk;
        bpk[0] = to_bf(b0.x); bpk[1] = to_bf(b0.y);
        bpk[2] = to_bf(b0.z); bpk[3] = to_bf(b0.w);
        bpk[4] = to_bf(b1.x); bpk[5] = to_bf(b1.y);
        bpk[6] = to_bf(b1.z); bpk[7] = to_bf(b1.w);
        *(v8bf*)&Bs[ar][ac] = bpk;

        __syncthreads();

        // A frag: lane m+16h, elem j -> K = 16h + j
        v16bf af, bf0, bf1;
        #pragma unroll
        for (int j = 0; j < 16; ++j) {
            int ka = 16 * hf + j;
            af[j] = As[wm * 16 + q16][ka];
        }
        // B frag: lane n+16h, elem j -> K = 16h + j  (col n = row of W)
        #pragma unroll
        for (int j = 0; j < 16; ++j) {
            int kb = 16 * hf + j;
            bf0[j] = Bs[wn * 32 + q16][kb];
            bf1[j] = Bs[wn * 32 + 16 + q16][kb];
        }
        acc0 = __builtin_amdgcn_wmma_f32_16x16x32_bf16(false, af, false, bf0,
                                                       (short)0, acc0, false, false);
        acc1 = __builtin_amdgcn_wmma_f32_16x16x32_bf16(false, af, false, bf1,
                                                       (short)0, acc1, false, false);
        __syncthreads();
    }

    // D layout: reg r, lane n+16h -> row 8h+r, col n
    const int n0 = tn + wn * 32 + q16;
    const int n1 = n0 + 16;
    const float b0 = bias ? bias[n0] : 0.f;
    const float b1 = bias ? bias[n1] : 0.f;
    #pragma unroll
    for (int r = 0; r < 8; ++r) {
        int m = tm + wm * 16 + 8 * hf + r;
        if (m < M) {
            float v0 = acc0[r] + b0;
            float v1 = acc1[r] + b1;
            if (relu) { v0 = fmaxf(v0, 0.f); v1 = fmaxf(v1, 0.f); }
            size_t o0 = (size_t)m * N + n0;
            size_t o1 = (size_t)m * N + n1;
            if (res) { v0 += res[o0]; v1 += res[o1]; }
            Cout[o0] = v0;
            Cout[o1] = v1;
        }
    }
}

// ===========================================================================
// Flash attention with the S^T trick.
//   S^T = K Q^T  (A = K tile rows, B = Q rows; both contiguous LDS/reg reads)
//   S^T D-layout: lane n+16h = query col n, key rows 8h+r
//     == A-fragment layout of P for O = P V  -> P transpose is register-only.
//   Per-lane online softmax (1 query/lane): in-register reductions + 1
//   shfl_xor(16). V staged transposed so both V B-frags are ds_load_b128.
// ===========================================================================
#define FA_WAVES 8
#define KB 32

__global__ __launch_bounds__(256) void flash_attn_kernel(
    const float* __restrict__ Qp, const float* __restrict__ Kv,
    const float* __restrict__ Vg, float* __restrict__ Op)
{
    __shared__ alignas(16) __bf16 Kt[KB][DH + 8];    // [key][d]
    __shared__ alignas(16) __bf16 VtT[DH][KB + 8];   // [d][key] (transposed)

    const int tid  = threadIdx.x;
    const int lane = tid & 31;
    const int wave = tid >> 5;
    const int hf   = lane >> 4;
    const int q16  = lane & 15;
    const int b    = blockIdx.y >> 3;
    const int head = blockIdx.y & 7;
    const int qt   = blockIdx.x * (FA_WAVES * 16) + wave * 16;
    const float scale = 0.17677669529663688f;   // 1/sqrt(32), folded into Q

    // Q fragment in B layout: lane n+16h, elem j -> Q[qt+n][16h+j]
    v16bf qf;
    {
        const float* qrow = Qp + ((size_t)b * LQ + qt + q16) * CDIM
                               + head * DH + 16 * hf;
        #pragma unroll
        for (int j = 0; j < 16; ++j) qf[j] = to_bf(qrow[j] * scale);
    }

    // staging coords: one float4 of K and V per thread per iteration
    const int kr = tid >> 3;             // 0..31 (key within tile)
    const int kc = (tid & 7) * 4;        // 0..28 (channel)
    const size_t kvbase = ((size_t)b * LQ + kr) * CDIM + head * DH + kc;

    v8f o0 = {0.f,0.f,0.f,0.f,0.f,0.f,0.f,0.f};
    v8f o1 = {0.f,0.f,0.f,0.f,0.f,0.f,0.f,0.f};
    float m_run = -3.0e38f;
    float l_run = 0.f;

    for (int kv = 0; kv < LQ; kv += KB) {
        // cooperative stage (shared by all 8 waves: same b,head)
        {
            const size_t o = kvbase + (size_t)kv * CDIM;
            float4 kk = *(const float4*)(Kv + o);
            float4 vv = *(const float4*)(Vg + o);
            if (kv + KB < LQ) {
                __builtin_prefetch(Kv + o + (size_t)KB * CDIM, 0, 1);
                __builtin_prefetch(Vg + o + (size_t)KB * CDIM, 0, 1);
            }
            v4bf kp;
            kp[0] = to_bf(kk.x); kp[1] = to_bf(kk.y);
            kp[2] = to_bf(kk.z); kp[3] = to_bf(kk.w);
            *(v4bf*)&Kt[kr][kc] = kp;
            VtT[kc + 0][kr] = to_bf(vv.x);
            VtT[kc + 1][kr] = to_bf(vv.y);
            VtT[kc + 2][kr] = to_bf(vv.z);
            VtT[kc + 3][kr] = to_bf(vv.w);
        }
        __syncthreads();

        // S^T = K Q^T: A-frag = K rows (contiguous-K pattern, 2x b128 each)
        v16bf af0, af1;
        #pragma unroll
        for (int j = 0; j < 16; ++j) {
            int d = 16 * hf + j;
            af0[j] = Kt[q16][d];
            af1[j] = Kt[16 + q16][d];
        }
        v8f z = {0.f,0.f,0.f,0.f,0.f,0.f,0.f,0.f};
        v8f s0 = __builtin_amdgcn_wmma_f32_16x16x32_bf16(false, af0, false, qf,
                                                         (short)0, z, false, false);
        v8f s1 = __builtin_amdgcn_wmma_f32_16x16x32_bf16(false, af1, false, qf,
                                                         (short)0, z, false, false);
        // lane n+16h now holds: query n, keys 8h+r (s0) and 16+8h+r (s1)

        // per-lane online softmax for this lane's query
        float mx = s0[0];
        #pragma unroll
        for (int r = 1; r < 8; ++r) mx = fmaxf(mx, s0[r]);
        #pragma unroll
        for (int r = 0; r < 8; ++r) mx = fmaxf(mx, s1[r]);
        mx = fmaxf(mx, __shfl_xor(mx, 16));
        const float mnew = fmaxf(m_run, mx);
        float p0[8], p1[8];
        float sum = 0.f;
        #pragma unroll
        for (int r = 0; r < 8; ++r) { p0[r] = __expf(s0[r] - mnew); sum += p0[r]; }
        #pragma unroll
        for (int r = 0; r < 8; ++r) { p1[r] = __expf(s1[r] - mnew); sum += p1[r]; }
        sum += __shfl_xor(sum, 16);
        const float corr = __expf(m_run - mnew);
        l_run = l_run * corr + sum;
        m_run = mnew;

        // P A-fragment directly from registers (layout coincidence)
        v16bf pf;
        #pragma unroll
        for (int j = 0; j < 8; ++j) {
            pf[j]     = to_bf(p0[j]);
            pf[j + 8] = to_bf(p1[j]);
        }

        // rescale O rows (row query 8h+r needs corr of query 8h+r)
        #pragma unroll
        for (int r = 0; r < 8; ++r) {
            float c = __shfl(corr, 8 * hf + r);
            o0[r] *= c;
            o1[r] *= c;
        }

        // V B-frags from transposed LDS: lane n+16h, elem j -> V[j<8 ? 8h+j : 8+8h+j][16t+n]
        v16bf vf0, vf1;
        #pragma unroll
        for (int j = 0; j < 16; ++j) {
            int d = (j < 8) ? (j + 8 * hf) : (j + 8 + 8 * hf);
            vf0[j] = VtT[q16][d];
            vf1[j] = VtT[16 + q16][d];
        }
        o0 = __builtin_amdgcn_wmma_f32_16x16x32_bf16(false, pf, false, vf0,
                                                     (short)0, o0, false, false);
        o1 = __builtin_amdgcn_wmma_f32_16x16x32_bf16(false, pf, false, vf1,
                                                     (short)0, o1, false, false);
        __syncthreads();
    }

    #pragma unroll
    for (int r = 0; r < 8; ++r) {
        float inv = 1.f / __shfl(l_run, 8 * hf + r);
        size_t o = ((size_t)b * LQ + qt + 8 * hf + r) * CDIM + head * DH;
        Op[o + q16]      = o0[r] * inv;
        Op[o + 16 + q16] = o1[r] * inv;
    }
}

// ===========================================================================
// LayerNorm over C=256: one wave per row, lane owns 8 contiguous channels
// (float4-pair loads/stores), butterfly reductions.
// ===========================================================================
__global__ __launch_bounds__(256) void layernorm_kernel(
    const float* __restrict__ x, const float* __restrict__ g,
    const float* __restrict__ bta, float* __restrict__ y, int rows)
{
    const int wave = threadIdx.x >> 5, lane = threadIdx.x & 31;
    const int row = blockIdx.x * 8 + wave;
    if (row >= rows) return;
    const int c0 = lane * 8;
    const float4* xr = (const float4*)(x + (size_t)row * CDIM + c0);
    float4 u0 = xr[0], u1 = xr[1];
    float s = u0.x + u0.y + u0.z + u0.w + u1.x + u1.y + u1.z + u1.w;
    s += __shfl_xor(s, 1);  s += __shfl_xor(s, 2);  s += __shfl_xor(s, 4);
    s += __shfl_xor(s, 8);  s += __shfl_xor(s, 16);
    const float mean = s * (1.f / 256.f);
    float var = 0.f;
    {
        float d;
        d = u0.x - mean; var += d * d;   d = u0.y - mean; var += d * d;
        d = u0.z - mean; var += d * d;   d = u0.w - mean; var += d * d;
        d = u1.x - mean; var += d * d;   d = u1.y - mean; var += d * d;
        d = u1.z - mean; var += d * d;   d = u1.w - mean; var += d * d;
    }
    var += __shfl_xor(var, 1);  var += __shfl_xor(var, 2);  var += __shfl_xor(var, 4);
    var += __shfl_xor(var, 8);  var += __shfl_xor(var, 16);
    const float inv = rsqrtf(var * (1.f / 256.f) + 1e-5f);
    const float4* gg = (const float4*)(g + c0);
    const float4* bb = (const float4*)(bta + c0);
    float4 g0 = gg[0], g1 = gg[1], b0 = bb[0], b1 = bb[1];
    float4 r0, r1;
    r0.x = (u0.x - mean) * inv * g0.x + b0.x;
    r0.y = (u0.y - mean) * inv * g0.y + b0.y;
    r0.z = (u0.z - mean) * inv * g0.z + b0.z;
    r0.w = (u0.w - mean) * inv * g0.w + b0.w;
    r1.x = (u1.x - mean) * inv * g1.x + b1.x;
    r1.y = (u1.y - mean) * inv * g1.y + b1.y;
    r1.z = (u1.z - mean) * inv * g1.z + b1.z;
    r1.w = (u1.w - mean) * inv * g1.w + b1.w;
    float4* yr = (float4*)(y + (size_t)row * CDIM + c0);
    yr[0] = r0; yr[1] = r1;
}

// ===========================================================================
// MSDeformAttn sampling core: one wave per (b, query, head); lane = channel.
// 16-way attention softmax in-lane + shuffle broadcast; bilinear gathers are
// 128B coalesced rows of `value`.
// ===========================================================================
__global__ __launch_bounds__(256) void deform_kernel(
    const float* __restrict__ value, const float* __restrict__ off,
    const float* __restrict__ awl, const int* __restrict__ shapes,
    const int* __restrict__ starts, const int* __restrict__ hptr,
    const int* __restrict__ wptr, float* __restrict__ samp)
{
    const int wave = threadIdx.x >> 5, lane = threadIdx.x & 31;
    const int q = blockIdx.x * 8 + wave;
    const int bh = blockIdx.y;
    const int b = bh >> 3, head = bh & 7;
    const size_t row = (size_t)b * LQ + q;

    // softmax over the 16 (level,point) logits of this head
    float lg = -3.0e38f;
    if (lane < 16) lg = awl[row * (NHEADS * 16) + head * 16 + lane];
    float mx = lg;
    mx = fmaxf(mx, __shfl_xor(mx, 1));  mx = fmaxf(mx, __shfl_xor(mx, 2));
    mx = fmaxf(mx, __shfl_xor(mx, 4));  mx = fmaxf(mx, __shfl_xor(mx, 8));
    mx = fmaxf(mx, __shfl_xor(mx, 16));
    float e = (lane < 16) ? __expf(lg - mx) : 0.f;
    float ssum = e;
    ssum += __shfl_xor(ssum, 1);  ssum += __shfl_xor(ssum, 2);
    ssum += __shfl_xor(ssum, 4);  ssum += __shfl_xor(ssum, 8);
    ssum += __shfl_xor(ssum, 16);
    const float wn = e / ssum;

    const int gw = wptr[0], gh = hptr[0];
    const float gx = ((float)(q % gw) + 0.5f) / (float)gw;
    const float gy = ((float)(q / gw) + 0.5f) / (float)gh;

    float acc = 0.f;
    #pragma unroll
    for (int l = 0; l < NLVL; ++l) {
        const int Hl = shapes[2 * l], Wl = shapes[2 * l + 1];
        const int st = starts[l];
        #pragma unroll
        for (int p = 0; p < 4; ++p) {
            const int oidx = ((head * NLVL + l) * 4 + p) * 2;
            const float ox = off[row * CDIM + oidx];
            const float oy = off[row * CDIM + oidx + 1];
            // xl = (gx + ox/Wl)*Wl - 0.5
            const float xl = gx * (float)Wl + ox - 0.5f;
            const float yl = gy * (float)Hl + oy - 0.5f;
            const float x0f = floorf(xl), y0f = floorf(yl);
            const float fx = xl - x0f, fy = yl - y0f;
            const int x0 = (int)x0f, y0 = (int)y0f;
            const float a = __shfl(wn, l * 4 + p);
            float pacc = 0.f;
            #pragma unroll
            for (int dy = 0; dy < 2; ++dy) {
                #pragma unroll
                for (int dx = 0; dx < 2; ++dx) {
                    const int xi = x0 + dx, yi = y0 + dy;
                    if (xi >= 0 && xi < Wl && yi >= 0 && yi < Hl) {
                        const float wgt = (dx ? fx : 1.f - fx) * (dy ? fy : 1.f - fy);
                        const size_t vi =
                            ((size_t)b * LEN_IN + st + yi * Wl + xi) * CDIM
                            + head * DH + lane;
                        pacc += value[vi] * wgt;
                    }
                }
            }
            acc += pacc * a;
        }
    }
    samp[row * CDIM + head * DH + lane] = acc;
}

// ===========================================================================
// Orchestration
// ===========================================================================
extern "C" void kernel_launch(void* const* d_in, const int* in_sizes, int n_in,
                              void* d_out, int out_size, void* d_ws, size_t ws_size,
                              hipStream_t stream)
{
    (void)in_sizes; (void)n_in; (void)out_size; (void)ws_size;
    const float* tgt   = (const float*)d_in[0];
    const float* qp    = (const float*)d_in[1];
    const float* src   = (const float*)d_in[2];
    const float* wq    = (const float*)d_in[3];  const float* bq    = (const float*)d_in[4];
    const float* wk    = (const float*)d_in[5];  const float* bk    = (const float*)d_in[6];
    const float* wv    = (const float*)d_in[7];  const float* bv    = (const float*)d_in[8];
    const float* wo    = (const float*)d_in[9];  const float* bo    = (const float*)d_in[10];
    const float* ln2g  = (const float*)d_in[11]; const float* ln2b  = (const float*)d_in[12];
    const float* woff  = (const float*)d_in[13]; const float* boff  = (const float*)d_in[14];
    const float* wattn = (const float*)d_in[15]; const float* battn = (const float*)d_in[16];
    const float* wval  = (const float*)d_in[17]; const float* bval  = (const float*)d_in[18];
    const float* wcout = (const float*)d_in[19]; const float* bcout = (const float*)d_in[20];
    const float* ln1g  = (const float*)d_in[21]; const float* ln1b  = (const float*)d_in[22];
    const float* w1    = (const float*)d_in[23]; const float* b1    = (const float*)d_in[24];
    const float* w2    = (const float*)d_in[25]; const float* b2    = (const float*)d_in[26];
    const float* ln3g  = (const float*)d_in[27]; const float* ln3b  = (const float*)d_in[28];
    const int*   shapes = (const int*)d_in[29];
    const int*   starts = (const int*)d_in[30];
    const int*   hptr   = (const int*)d_in[31];
    const int*   wptr   = (const int*)d_in[32];
    float* out = (float*)d_out;

    const int M = BDIM * LQ;                // 4608
    const size_t SZ = (size_t)M * CDIM;
    float* f    = (float*)d_ws;
    float* Q    = f;
    float* Kp   = Q + SZ;
    float* Vp   = Kp + SZ;
    float* attn = Vp + SZ;
    float* x1   = attn + SZ;
    float* t    = x1 + SZ;
    float* val  = t + SZ;
    float* off  = val + (size_t)6144 * CDIM;       // 6120 rows, padded
    float* awl  = off + SZ;
    float* samp = awl + (size_t)M * 128;
    float* x2   = samp + SZ;
    float* t2   = x2 + SZ;
    float* ffh  = t2 + SZ;
    float* x3   = ffh + (size_t)M * DFF;

    const dim3 blk(256);

    // --- self attention (q = k = tgt+pos, v = tgt) ---
    gemm_bf16_kernel<true ><<<dim3(M/64, CDIM/64), blk, 0, stream>>>(tgt, qp, wq, bq, nullptr, Q,  M, CDIM, CDIM, 0);
    gemm_bf16_kernel<true ><<<dim3(M/64, CDIM/64), blk, 0, stream>>>(tgt, qp, wk, bk, nullptr, Kp, M, CDIM, CDIM, 0);
    gemm_bf16_kernel<false><<<dim3(M/64, CDIM/64), blk, 0, stream>>>(tgt, nullptr, wv, bv, nullptr, Vp, M, CDIM, CDIM, 0);
    flash_attn_kernel<<<dim3(LQ/128, BDIM*NHEADS), blk, 0, stream>>>(Q, Kp, Vp, attn);
    gemm_bf16_kernel<false><<<dim3(M/64, CDIM/64), blk, 0, stream>>>(attn, nullptr, wo, bo, tgt, x1, M, CDIM, CDIM, 0);
    layernorm_kernel<<<dim3(M/8), blk, 0, stream>>>(x1, ln2g, ln2b, t, M);

    // --- MSDeformAttn cross attention ---
    const int Mv = BDIM * LEN_IN;           // 6120
    gemm_bf16_kernel<false><<<dim3((Mv+63)/64, CDIM/64), blk, 0, stream>>>(src, nullptr, wval, bval, nullptr, val, Mv, CDIM, CDIM, 0);
    gemm_bf16_kernel<true ><<<dim3(M/64, CDIM/64), blk, 0, stream>>>(t, qp, woff,  boff,  nullptr, off, M, 256, CDIM, 0);
    gemm_bf16_kernel<true ><<<dim3(M/64, 128/64),  blk, 0, stream>>>(t, qp, wattn, battn, nullptr, awl, M, 128, CDIM, 0);
    deform_kernel<<<dim3(LQ/8, BDIM*NHEADS), blk, 0, stream>>>(val, off, awl, shapes, starts, hptr, wptr, samp);
    gemm_bf16_kernel<false><<<dim3(M/64, CDIM/64), blk, 0, stream>>>(samp, nullptr, wcout, bcout, t, x2, M, CDIM, CDIM, 0);
    layernorm_kernel<<<dim3(M/8), blk, 0, stream>>>(x2, ln1g, ln1b, t2, M);

    // --- FFN ---
    gemm_bf16_kernel<false><<<dim3(M/64, DFF/64),  blk, 0, stream>>>(t2, nullptr, w1, b1, nullptr, ffh, M, DFF, CDIM, 1);
    gemm_bf16_kernel<false><<<dim3(M/64, CDIM/64), blk, 0, stream>>>(ffh, nullptr, w2, b2, t2, x3, M, CDIM, DFF, 0);
    layernorm_kernel<<<dim3(M/8), blk, 0, stream>>>(x3, ln3g, ln3b, out, M);
}
